// MyModel_87522843559084
// MI455X (gfx1250) — matrix operand order repaired and checked by reading, not compiled, not run on hardware
//
#include <hip/hip_runtime.h>

#define BATCH   4096
#define SEQ     200
#define HID     128
#define THREE_H 384
#define BM      32      // batch rows per workgroup (two 16-row M tiles) == wave32 size
#define NTHREADS 256

typedef __attribute__((ext_vector_type(16))) __bf16 v16bf;
typedef __attribute__((ext_vector_type(8)))  float  v8f;

union FragU { v16bf v; uint4 q[2]; };

__device__ __forceinline__ unsigned short f2bf(float f) {
    unsigned u = __float_as_uint(f);
    unsigned r = u + 0x7FFFu + ((u >> 16) & 1u);   // round-to-nearest-even
    return (unsigned short)(r >> 16);
}

__device__ __forceinline__ float sigmoidf_fast(float x) {
    return 1.0f / (1.0f + __expf(-x));
}

__device__ __forceinline__ unsigned ballot32(bool p) {
#if __has_builtin(__builtin_amdgcn_ballot_w32)
    return __builtin_amdgcn_ballot_w32(p);
#else
    return (unsigned)__ballot(p);
#endif
}

// A-fragment: 16x32 bf16 tile from LDS [M][K] row-major.
// lane L: m = mt*16 + (L&15); VGPR0-3 hold K = khalf..khalf+7, VGPR4-7 hold K = 16+khalf..
__device__ __forceinline__ v16bf loadA(const unsigned short* s, int mt, int kb, int lane) {
    int m     = mt * 16 + (lane & 15);
    int khalf = (lane < 16) ? 0 : 8;
    const unsigned short* p = s + m * HID + kb * 32 + khalf;
    FragU f;
    f.q[0] = *(const uint4*)(p);
    f.q[1] = *(const uint4*)(p + 16);
    return f.v;
}

// B-fragment: 32x16 bf16 tile from LDS weights stored transposed [N][K].
// lane L: n = gate*128 + wv*16 + (L&15); VGPR0-3 hold K = kq..kq+7, VGPR4-7 hold K = 16+kq.. (kq = 0 / 8)
__device__ __forceinline__ v16bf loadB(const unsigned short* sw, int gate, int wv, int kb, int lane) {
    int n  = gate * HID + wv * 16 + (lane & 15);
    int kq = (lane < 16) ? 0 : 8;
    const unsigned short* p = sw + n * HID + kb * 32 + kq;
    FragU f;
    f.q[0] = *(const uint4*)(p);
    f.q[1] = *(const uint4*)(p + 16);
    return f.v;
}

#define WMMA_BF16(A, B, C) \
    __builtin_amdgcn_wmma_f32_16x16x32_bf16(false, (A), false, (B), (short)0, (C), false, false)

// Gather BM embedding rows (f32 -> bf16) into LDS x-tile. 512 chunks of 8 elems, 2 per thread.
__device__ __forceinline__ void gather_rows(const float* __restrict__ emb,
                                            const int* tok, unsigned short* dst, int tid) {
#pragma unroll
    for (int i = 0; i < 2; ++i) {
        int c   = tid + i * NTHREADS;       // chunk id
        int row = c >> 4;                   // 16 chunks per 128-elem row
        int off = (c & 15) << 3;
        const float* src = emb + (size_t)tok[row] * HID + off;
        float4 a = *(const float4*)(src);
        float4 b = *(const float4*)(src + 4);
        union { unsigned short u[8]; uint4 q; } pk;
        pk.u[0] = f2bf(a.x); pk.u[1] = f2bf(a.y); pk.u[2] = f2bf(a.z); pk.u[3] = f2bf(a.w);
        pk.u[4] = f2bf(b.x); pk.u[5] = f2bf(b.y); pk.u[6] = f2bf(b.z); pk.u[7] = f2bf(b.w);
        *(uint4*)(dst + row * HID + off) = pk.q;
    }
}

__global__ __launch_bounds__(NTHREADS, 1)
void gru_fused_kernel(const int*   __restrict__ tokens,   // (BATCH, SEQ) int32
                      const float* __restrict__ Wk,       // (HID, 3H) row-major
                      const float* __restrict__ Wr,       // (HID, 3H) row-major
                      const float* __restrict__ emb,      // (VOCAB, HID)
                      const float* __restrict__ bias,     // (2, 3H)
                      float*       __restrict__ out)      // (BATCH, HID)
{
    // LDS: weights transposed [N=384][K=128] bf16, x/h tiles [32][128] bf16, token+mask buffers
    __shared__ __align__(16) unsigned short s_w[2][THREE_H * HID];  // 2 x 96KB
    __shared__ __align__(16) unsigned short s_x[BM * HID];          // 8KB
    __shared__ __align__(16) unsigned short s_h[BM * HID];          // 8KB
    __shared__ int      s_tok[2][BM];
    __shared__ unsigned s_msk[2];

    const int tid   = threadIdx.x;
    const int lane  = tid & 31;
    const int wv    = tid >> 5;                 // 0..7: hidden 16-col slice per wave
    const int bbase = blockIdx.x * BM;

    // ---- one-time: load + transpose + convert weights into LDS ----
    for (int mat = 0; mat < 2; ++mat) {
        const float* W = mat ? Wr : Wk;
        unsigned short* dw = &s_w[mat][0];
        for (int e = tid; e < THREE_H * HID; e += NTHREADS) {
            int k = e / THREE_H;                // consecutive tids -> consecutive n: coalesced read
            int n = e % THREE_H;
            dw[n * HID + k] = f2bf(W[e]);
        }
    }

    // tokens + mask for t=0; zero h tile
    {
        int tok0 = 0;
        if (tid < BM) {
            tok0 = tokens[(size_t)(bbase + tid) * SEQ + 0];
            s_tok[0][tid] = tok0;
        }
        if (wv == 0) {                          // BM == 32 == wave size
            unsigned mk = ballot32(tok0 != 0);
            if (lane == 0) s_msk[0] = mk;
        }
    }
    for (int e = tid; e < (BM * HID) / 8; e += NTHREADS)
        ((uint4*)s_h)[e] = make_uint4(0u, 0u, 0u, 0u);
    __syncthreads();

    gather_rows(emb, &s_tok[0][0], s_x, tid);   // x(0)

    // ---- pin recurrent-matrix B fragments in VGPRs for the whole scan (96 VGPRs) ----
    const unsigned short* s_w0 = &s_w[0][0];
    const unsigned short* s_w1 = &s_w[1][0];
    v16bf Rz[4], Rr[4], Rh[4];
#pragma unroll
    for (int kb = 0; kb < 4; ++kb) {
        Rz[kb] = loadB(s_w1, 0, wv, kb, lane);
        Rr[kb] = loadB(s_w1, 1, wv, kb, lane);
        Rh[kb] = loadB(s_w1, 2, wv, kb, lane);
    }
    __syncthreads();

    // per-lane bias init (accumulator seeds)
    const int ncol = wv * 16 + (lane & 15);     // hidden column 0..127
    const float bz  = bias[ncol]           + bias[THREE_H + ncol];
    const float br  = bias[HID + ncol]     + bias[THREE_H + HID + ncol];
    const float bxh = bias[2 * HID + ncol];                 // input-side only
    const float bhh = bias[THREE_H + 2 * HID + ncol];       // rec-side only

    float hstate[2][8];
#pragma unroll
    for (int mt = 0; mt < 2; ++mt)
#pragma unroll
        for (int r = 0; r < 8; ++r) hstate[mt][r] = 0.0f;

    for (int t = 0; t < SEQ; ++t) {
        const int cur = t & 1, nxt = cur ^ 1;

        // ---- phase 1: prefetch next tokens/mask + 48 WMMAs ----
        {
            int tokn = 0;
            if (tid < BM) {
                tokn = (t + 1 < SEQ) ? tokens[(size_t)(bbase + tid) * SEQ + (t + 1)] : 0;
                s_tok[nxt][tid] = tokn;
            }
            if (wv == 0) {
                unsigned mk = ballot32(tokn != 0);
                if (lane == 0) s_msk[nxt] = mk;
            }
        }

        v8f az[2], ar[2], axh[2], ahh[2];
#pragma unroll
        for (int mt = 0; mt < 2; ++mt)
#pragma unroll
            for (int i = 0; i < 8; ++i) {
                az[mt][i] = bz; ar[mt][i] = br; axh[mt][i] = bxh; ahh[mt][i] = bhh;
            }

        // h @ Wr  (z, r, hh) — register-resident B fragments, critical path first
#pragma unroll
        for (int kb = 0; kb < 4; ++kb) {
            v16bf a0 = loadA(s_h, 0, kb, lane);
            v16bf a1 = loadA(s_h, 1, kb, lane);
            az[0]  = WMMA_BF16(a0, Rz[kb], az[0]);   az[1]  = WMMA_BF16(a1, Rz[kb], az[1]);
            ar[0]  = WMMA_BF16(a0, Rr[kb], ar[0]);   ar[1]  = WMMA_BF16(a1, Rr[kb], ar[1]);
            ahh[0] = WMMA_BF16(a0, Rh[kb], ahh[0]);  ahh[1] = WMMA_BF16(a1, Rh[kb], ahh[1]);
        }
        // x @ Wk  (z, r, xh) — B fragments streamed from LDS
#pragma unroll
        for (int kb = 0; kb < 4; ++kb) {
            v16bf a0 = loadA(s_x, 0, kb, lane);
            v16bf a1 = loadA(s_x, 1, kb, lane);
            v16bf b;
            b = loadB(s_w0, 0, wv, kb, lane);
            az[0]  = WMMA_BF16(a0, b, az[0]);   az[1]  = WMMA_BF16(a1, b, az[1]);
            b = loadB(s_w0, 1, wv, kb, lane);
            ar[0]  = WMMA_BF16(a0, b, ar[0]);   ar[1]  = WMMA_BF16(a1, b, ar[1]);
            b = loadB(s_w0, 2, wv, kb, lane);
            axh[0] = WMMA_BF16(a0, b, axh[0]);  axh[1] = WMMA_BF16(a1, b, axh[1]);
        }

        __syncthreads();   // all tile reads done; s_tok/s_msk[nxt] visible

        // ---- phase 2: gates in registers, write bf16 h tile, gather x(t+1) ----
        const unsigned mk = s_msk[cur];
#pragma unroll
        for (int mt = 0; mt < 2; ++mt) {
#pragma unroll
            for (int r = 0; r < 8; ++r) {
                int m = ((lane >> 4) << 3) + r;                 // row within M tile
                bool msk = (mk >> (mt * 16 + m)) & 1u;
                float z  = sigmoidf_fast(az[mt][r]);
                float rr = sigmoidf_fast(ar[mt][r]);
                float hh = tanhf(axh[mt][r] + rr * ahh[mt][r]);
                float hn = z * hstate[mt][r] + (1.0f - z) * hh;
                hstate[mt][r] = msk ? hn : hstate[mt][r];
                s_h[(mt * 16 + m) * HID + ncol] = f2bf(hstate[mt][r]);
            }
        }
        if (t + 1 < SEQ)
            gather_rows(emb, &s_tok[nxt][0], s_x, tid);
        __syncthreads();
    }

    // ---- epilogue: write f32 h_final ----
#pragma unroll
    for (int mt = 0; mt < 2; ++mt)
#pragma unroll
        for (int r = 0; r < 8; ++r) {
            int m = ((lane >> 4) << 3) + r;
            out[(size_t)(bbase + mt * 16 + m) * HID + ncol] = hstate[mt][r];
        }
}

extern "C" void kernel_launch(void* const* d_in, const int* in_sizes, int n_in,
                              void* d_out, int out_size, void* d_ws, size_t ws_size,
                              hipStream_t stream) {
    const int*   tokens = (const int*)  d_in[0];   // (BATCH, SEQ) int32
    const float* emb    = (const float*)d_in[1];   // (VOCAB, HID)
    const float* Wk     = (const float*)d_in[2];   // (HID, 3H)
    const float* Wr     = (const float*)d_in[3];   // (HID, 3H)
    const float* bias   = (const float*)d_in[4];   // (2, 3H)
    float* out = (float*)d_out;                    // (BATCH, HID)

    dim3 grid(BATCH / BM);
    dim3 block(NTHREADS);
    hipLaunchKernelGGL(gru_fused_kernel, grid, block, 0, stream,
                       tokens, Wk, Wr, emb, bias, out);
}
